// GATv2_LCG_15839839387882
// MI455X (gfx1250) — matrix operand order repaired and checked by reading, not compiled, not run on hardware
//
#include <hip/hip_runtime.h>

#define DIMN 128
#define NITER 4
#define NLAYER 2
#define LSZ 100000
#define CSZ 40000
#define NEDG 400000
#define NEG_SLOPE 0.2f

typedef _Float16 v16h __attribute__((ext_vector_type(16)));
typedef _Float16 v8h  __attribute__((ext_vector_type(8)));
typedef _Float16 v4h  __attribute__((ext_vector_type(4)));
typedef float    v8f  __attribute__((ext_vector_type(8)));

union U16h { v16h v; v8h h[2]; };

static_assert(LSZ % 16 == 0 && CSZ % 16 == 0, "row tiles must divide");

__device__ __forceinline__ void atomAddF(float* p, float v) {
  __hip_atomic_fetch_add(p, v, __ATOMIC_RELAXED, __HIP_MEMORY_SCOPE_AGENT);
}
// order-preserving float <-> uint map for atomicMax on floats
__device__ __forceinline__ unsigned encOrd(float f) {
  unsigned u = __float_as_uint(f);
  return (u & 0x80000000u) ? ~u : (u | 0x80000000u);
}
__device__ __forceinline__ float decOrd(unsigned u) {
  return __uint_as_float((u & 0x80000000u) ? (u ^ 0x80000000u) : ~u);
}

// ---------------------------------------------------------------------------
// WMMA GEMM: out[M,128] = act( A[M,K](f16) @ Wt[K->] + bias ), Wt stored [128,K]
// Block = 256 threads = 8 waves; block handles 16 rows, wave w -> cols 16w..16w+15.
// Per K=32 chunk each lane loads A: two contiguous v8h at k = 16*hi (+8),
// B: one contiguous v16h at k = 16*hi  (contiguous k = 16*hi + s order, shared by A and B).
// ---------------------------------------------------------------------------
__global__ void __launch_bounds__(256)
gemm_wmma_f16(const _Float16* __restrict__ A, int aStride, int rowXor,
              const _Float16* __restrict__ Bt,
              const float* __restrict__ bias,
              int K, int relu,
              float* __restrict__ outF, int ofStride,
              _Float16* __restrict__ outH, int ohStride)
{
  const int lane = threadIdx.x & 31;
  const int wave = threadIdx.x >> 5;     // col tile 0..7
  const int hi   = lane >> 4;            // half-wave
  const int l15  = lane & 15;
  const int rtile = blockIdx.x;
  const int arow  = (rtile * 16 + l15) ^ rowXor;
  const int col   = wave * 16 + l15;

  const _Float16* ap = A  + (size_t)arow * aStride;
  const _Float16* bp = Bt + (size_t)col * K;

  v8f acc = {};
  const int nk = K >> 5;
  for (int kc = 0; kc < nk; ++kc) {
    const int ka = (kc << 5) + (hi << 4);
    const int kb = (kc << 5) + (hi << 4);
    U16h a, b;
    a.h[0] = *(const v8h*)(ap + ka);
    a.h[1] = *(const v8h*)(ap + ka + 8);
    b.v    = *(const v16h*)(bp + kb);
    acc = __builtin_amdgcn_wmma_f32_16x16x32_f16(
        false, a.v, false, b.v, (short)0, acc, false, false);
  }

  const float bv = bias[col];
  const int rbase = rtile * 16 + (hi << 3);
#pragma unroll
  for (int v = 0; v < 8; ++v) {
    float x = acc[v] + bv;
    if (relu) x = x > 0.0f ? x : 0.0f;
    if (outF) outF[(size_t)(rbase + v) * ofStride + col] = x;
    if (outH) outH[(size_t)(rbase + v) * ohStride + col] = (_Float16)x;
  }
}

// f32[M,128] (stride sS) -> f16 (stride dS), used to fill concat columns
__global__ void cvt_cat(const float* __restrict__ src, int sS,
                        _Float16* __restrict__ dst, int dS, int M)
{
  int t = blockIdx.x * blockDim.x + threadIdx.x;
  if (t >= M * (DIMN / 4)) return;
  int r  = t >> 5;
  int c4 = (t & 31) << 2;
  float4 v = *(const float4*)(src + (size_t)r * sS + c4);
  v4h h; h[0] = (_Float16)v.x; h[1] = (_Float16)v.y;
         h[2] = (_Float16)v.z; h[3] = (_Float16)v.w;
  *(v4h*)(dst + (size_t)r * dS + c4) = h;
}

// W[m][k][n] (f32) -> Wt[m][n][k] (f16), N == 128
__global__ void wt_cvt(const float* __restrict__ W, _Float16* __restrict__ Wt,
                       int nmat, int K)
{
  int idx = blockIdx.x * blockDim.x + threadIdx.x;
  int total = nmat * K * DIMN;
  if (idx >= total) return;
  int m = idx / (K * DIMN);
  int rem = idx - m * K * DIMN;
  int n = rem / K;
  int k = rem - n * K;
  Wt[idx] = (_Float16)W[(size_t)m * K * DIMN + (size_t)k * DIMN + n];
}

__global__ void copy_f4(const float* __restrict__ s, float* __restrict__ d, int n4)
{
  int i = blockIdx.x * blockDim.x + threadIdx.x;
  if (i < n4) ((float4*)d)[i] = ((const float4*)s)[i];
}

__global__ void zero_f4(float* __restrict__ p, int n4)
{
  int i = blockIdx.x * blockDim.x + threadIdx.x;
  if (i < n4) ((float4*)p)[i] = make_float4(0.f, 0.f, 0.f, 0.f);
}

// One wave per edge: attention logits + segment max (both keyed as reference)
__global__ void __launch_bounds__(256)
edge_logits(const int* __restrict__ lidx, const int* __restrict__ cidx,
            const float* __restrict__ lCur, const float* __restrict__ cCur,
            const float* __restrict__ lMsg, const float* __restrict__ cMsg,
            const float* __restrict__ cAtt, const float* __restrict__ lAtt,
            float* __restrict__ s1, float* __restrict__ s2,
            unsigned* __restrict__ m1, unsigned* __restrict__ m2)
{
  int wid = (blockIdx.x * blockDim.x + threadIdx.x) >> 5;
  if (wid >= NEDG) return;
  int lane = threadIdx.x & 31;
  int le = lidx[wid], ce = cidx[wid];

  float4 a, b;
  float d1 = 0.f, d2 = 0.f;
  a = ((const float4*)(cCur + (size_t)ce * DIMN))[lane];
  b = ((const float4*)cAtt)[lane];
  d1 += a.x * b.x + a.y * b.y + a.z * b.z + a.w * b.w;
  a = ((const float4*)(lMsg + (size_t)le * DIMN))[lane];
  b = ((const float4*)(cAtt + DIMN))[lane];
  d1 += a.x * b.x + a.y * b.y + a.z * b.z + a.w * b.w;
  a = ((const float4*)(lCur + (size_t)le * DIMN))[lane];
  b = ((const float4*)lAtt)[lane];
  d2 += a.x * b.x + a.y * b.y + a.z * b.z + a.w * b.w;
  a = ((const float4*)(cMsg + (size_t)ce * DIMN))[lane];
  b = ((const float4*)(lAtt + DIMN))[lane];
  d2 += a.x * b.x + a.y * b.y + a.z * b.z + a.w * b.w;

#pragma unroll
  for (int o = 16; o > 0; o >>= 1) {
    d1 += __shfl_xor(d1, o, 32);
    d2 += __shfl_xor(d2, o, 32);
  }
  if (lane == 0) {
    float v1 = d1 > 0.f ? d1 : NEG_SLOPE * d1;
    float v2 = d2 > 0.f ? d2 : NEG_SLOPE * d2;
    s1[wid] = v1; s2[wid] = v2;
    atomicMax(m1 + ce, encOrd(v1));   // both softmaxes segment on c_edge_index
    atomicMax(m2 + ce, encOrd(v2));   // (reference quirk preserved)
  }
}

// exp(logit - segmax) + segment sum
__global__ void edge_expsum(const int* __restrict__ cidx,
                            float* __restrict__ s1, float* __restrict__ s2,
                            const unsigned* __restrict__ m1, const unsigned* __restrict__ m2,
                            float* __restrict__ z1, float* __restrict__ z2)
{
  int e = blockIdx.x * blockDim.x + threadIdx.x;
  if (e >= NEDG) return;
  int ce = cidx[e];
  float e1 = expf(s1[e] - decOrd(m1[ce]));
  float e2 = expf(s2[e] - decOrd(m2[ce]));
  s1[e] = e1; s2[e] = e2;
  atomAddF(z1 + ce, e1);
  atomAddF(z2 + ce, e2);
}

// weighted scatter-add of messages (one wave per edge, float4 per lane)
__global__ void __launch_bounds__(256)
edge_scatter(const int* __restrict__ lidx, const int* __restrict__ cidx,
             const float* __restrict__ s1, const float* __restrict__ s2,
             const float* __restrict__ z1, const float* __restrict__ z2,
             const float* __restrict__ lMsg, const float* __restrict__ cMsg,
             float* __restrict__ l2cA, float* __restrict__ c2lA)
{
  int wid = (blockIdx.x * blockDim.x + threadIdx.x) >> 5;
  if (wid >= NEDG) return;
  int lane = threadIdx.x & 31;
  int le = lidx[wid], ce = cidx[wid];
  float w1 = s1[wid] / (z1[ce] + 1e-16f);
  float w2 = s2[wid] / (z2[ce] + 1e-16f);

  float4 mv = ((const float4*)(lMsg + (size_t)le * DIMN))[lane];
  float* d1 = l2cA + (size_t)ce * DIMN + lane * 4;
  atomAddF(d1 + 0, w1 * mv.x); atomAddF(d1 + 1, w1 * mv.y);
  atomAddF(d1 + 2, w1 * mv.z); atomAddF(d1 + 3, w1 * mv.w);

  float4 cv = ((const float4*)(cMsg + (size_t)ce * DIMN))[lane];
  float* d2 = c2lA + (size_t)le * DIMN + lane * 4;
  atomAddF(d2 + 0, w2 * cv.x); atomAddF(d2 + 1, w2 * cv.y);
  atomAddF(d2 + 2, w2 * cv.z); atomAddF(d2 + 3, w2 * cv.w);
}

// ---------------------------------------------------------------------------
extern "C" void kernel_launch(void* const* d_in, const int* in_sizes, int n_in,
                              void* d_out, int out_size, void* d_ws, size_t ws_size,
                              hipStream_t stream) {
  (void)in_sizes; (void)n_in; (void)out_size; (void)ws_size;

  const float* l_emb  = (const float*)d_in[0];
  const float* c_emb  = (const float*)d_in[1];
  const float* l2c_W  = (const float*)d_in[2];
  const float* l2c_b  = (const float*)d_in[3];
  const float* c2l_W  = (const float*)d_in[4];
  const float* c2l_b  = (const float*)d_in[5];
  const float* l2l_W  = (const float*)d_in[6];
  const float* l2l_b  = (const float*)d_in[7];
  const float* c_att  = (const float*)d_in[8];
  const float* l_att  = (const float*)d_in[9];
  const float* cu_W   = (const float*)d_in[10];
  const float* cu_b   = (const float*)d_in[11];
  const float* lu_W   = (const float*)d_in[12];
  const float* lu_b   = (const float*)d_in[13];
  const int*   l_idx  = (const int*)d_in[14];
  const int*   c_idx  = (const int*)d_in[15];

  float* out_l = (float*)d_out;                            // [5, LSZ, 128]
  float* out_c = out_l + (size_t)(NITER + 1) * LSZ * DIMN; // [5, CSZ, 128]

  // workspace bump allocator (256B aligned)
  char* ws = (char*)d_ws;
  size_t off = 0;
  auto walloc = [&](size_t bytes) -> void* {
    off = (off + 255) & ~(size_t)255;
    void* p = ws + off;
    off += bytes;
    return p;
  };
  _Float16* WtL2C = (_Float16*)walloc((size_t)NITER * NLAYER * DIMN * DIMN * 2);
  _Float16* WtC2L = (_Float16*)walloc((size_t)NITER * NLAYER * DIMN * DIMN * 2);
  _Float16* WtL2L = (_Float16*)walloc((size_t)NITER * NLAYER * DIMN * DIMN * 2);
  _Float16* WtCU  = (_Float16*)walloc((size_t)NITER * 2 * DIMN * DIMN * 2);
  _Float16* WtLU  = (_Float16*)walloc((size_t)NITER * 3 * DIMN * DIMN * 2);
  _Float16* l_cat = (_Float16*)walloc((size_t)LSZ * 3 * DIMN * 2); // [l | c2lA | l2l]
  _Float16* c_cat = (_Float16*)walloc((size_t)CSZ * 2 * DIMN * 2); // [c | l2cA]
  _Float16* l_hid = (_Float16*)walloc((size_t)LSZ * DIMN * 2);
  _Float16* lf_hid= (_Float16*)walloc((size_t)LSZ * DIMN * 2);
  _Float16* c_hid = (_Float16*)walloc((size_t)CSZ * DIMN * 2);
  float* l_msg = (float*)walloc((size_t)LSZ * DIMN * 4);
  float* c_msg = (float*)walloc((size_t)CSZ * DIMN * 4);
  float* s1 = (float*)walloc((size_t)NEDG * 4);
  float* s2 = (float*)walloc((size_t)NEDG * 4);
  // contiguous zero region: l2cA | c2lA | z1 | z2 | m1 | m2
  const size_t nzero = (size_t)CSZ * DIMN + (size_t)LSZ * DIMN + 4 * (size_t)CSZ;
  float* zbase = (float*)walloc(nzero * 4);
  float* l2cA = zbase;
  float* c2lA = l2cA + (size_t)CSZ * DIMN;
  float* z1   = c2lA + (size_t)LSZ * DIMN;
  float* z2   = z1 + CSZ;
  unsigned* m1 = (unsigned*)(z2 + CSZ);
  unsigned* m2 = m1 + CSZ;

  auto cdiv = [](int a, int b) { return (a + b - 1) / b; };
  auto gemm = [&](const _Float16* A, int aS, int rx, const _Float16* Bt,
                  const float* bias, int M, int K, int relu,
                  float* oF, int ofS, _Float16* oH, int ohS) {
    gemm_wmma_f16<<<M / 16, 256, 0, stream>>>(A, aS, rx, Bt, bias, K, relu,
                                              oF, ofS, oH, ohS);
  };

  // initial output slices
  copy_f4<<<cdiv(LSZ * DIMN / 4, 256), 256, 0, stream>>>(l_emb, out_l, LSZ * DIMN / 4);
  copy_f4<<<cdiv(CSZ * DIMN / 4, 256), 256, 0, stream>>>(c_emb, out_c, CSZ * DIMN / 4);

  // transpose+convert all weights once
  wt_cvt<<<cdiv(NITER * NLAYER * DIMN * DIMN, 256), 256, 0, stream>>>(l2c_W, WtL2C, NITER * NLAYER, DIMN);
  wt_cvt<<<cdiv(NITER * NLAYER * DIMN * DIMN, 256), 256, 0, stream>>>(c2l_W, WtC2L, NITER * NLAYER, DIMN);
  wt_cvt<<<cdiv(NITER * NLAYER * DIMN * DIMN, 256), 256, 0, stream>>>(l2l_W, WtL2L, NITER * NLAYER, DIMN);
  wt_cvt<<<cdiv(NITER * 2 * DIMN * DIMN, 256), 256, 0, stream>>>(cu_W, WtCU, NITER, 2 * DIMN);
  wt_cvt<<<cdiv(NITER * 3 * DIMN * DIMN, 256), 256, 0, stream>>>(lu_W, WtLU, NITER, 3 * DIMN);

  for (int i = 0; i < NITER; ++i) {
    const float* l_cur = out_l + (size_t)i * LSZ * DIMN;
    const float* c_cur = out_c + (size_t)i * CSZ * DIMN;
    float* l_next = out_l + (size_t)(i + 1) * LSZ * DIMN;
    float* c_next = out_c + (size_t)(i + 1) * CSZ * DIMN;
    const _Float16* wl2c0 = WtL2C + (size_t)(i * NLAYER + 0) * DIMN * DIMN;
    const _Float16* wl2c1 = WtL2C + (size_t)(i * NLAYER + 1) * DIMN * DIMN;
    const _Float16* wc2l0 = WtC2L + (size_t)(i * NLAYER + 0) * DIMN * DIMN;
    const _Float16* wc2l1 = WtC2L + (size_t)(i * NLAYER + 1) * DIMN * DIMN;
    const _Float16* wl2l0 = WtL2L + (size_t)(i * NLAYER + 0) * DIMN * DIMN;
    const _Float16* wl2l1 = WtL2L + (size_t)(i * NLAYER + 1) * DIMN * DIMN;

    // zero aggregation / softmax state (m encoded-uint 0 == below any real)
    zero_f4<<<cdiv((int)(nzero / 4), 256), 256, 0, stream>>>(zbase, (int)(nzero / 4));

    // fill cat cols 0..127 with current embeddings (f16)
    cvt_cat<<<cdiv(LSZ * 32, 256), 256, 0, stream>>>(l_cur, DIMN, l_cat, 3 * DIMN, LSZ);
    cvt_cat<<<cdiv(CSZ * 32, 256), 256, 0, stream>>>(c_cur, DIMN, c_cat, 2 * DIMN, CSZ);

    // MLPs (WMMA)
    gemm(l_cat, 3 * DIMN, 0, wl2c0, l2c_b + (i * NLAYER + 0) * DIMN, LSZ, DIMN, 1, nullptr, 0, l_hid, DIMN);
    gemm(l_hid, DIMN, 0, wl2c1, l2c_b + (i * NLAYER + 1) * DIMN, LSZ, DIMN, 0, l_msg, DIMN, nullptr, 0);
    gemm(c_cat, 2 * DIMN, 0, wc2l0, c2l_b + (i * NLAYER + 0) * DIMN, CSZ, DIMN, 1, nullptr, 0, c_hid, DIMN);
    gemm(c_hid, DIMN, 0, wc2l1, c2l_b + (i * NLAYER + 1) * DIMN, CSZ, DIMN, 0, c_msg, DIMN, nullptr, 0);
    gemm(l_cat, 3 * DIMN, 1, wl2l0, l2l_b + (i * NLAYER + 0) * DIMN, LSZ, DIMN, 1, nullptr, 0, lf_hid, DIMN);
    gemm(lf_hid, DIMN, 0, wl2l1, l2l_b + (i * NLAYER + 1) * DIMN, LSZ, DIMN, 0, nullptr, 0, l_cat + 2 * DIMN, 3 * DIMN);

    // attention + segment softmax + weighted aggregation
    edge_logits<<<cdiv(NEDG * 32, 256), 256, 0, stream>>>(
        l_idx, c_idx, l_cur, c_cur, l_msg, c_msg,
        c_att + (size_t)i * 2 * DIMN, l_att + (size_t)i * 2 * DIMN,
        s1, s2, m1, m2);
    edge_expsum<<<cdiv(NEDG, 256), 256, 0, stream>>>(c_idx, s1, s2, m1, m2, z1, z2);
    edge_scatter<<<cdiv(NEDG * 32, 256), 256, 0, stream>>>(
        l_idx, c_idx, s1, s2, z1, z2, l_msg, c_msg, l2cA, c2lA);

    // fill cat cols 128.. with aggregates (f16)
    cvt_cat<<<cdiv(CSZ * 32, 256), 256, 0, stream>>>(l2cA, DIMN, c_cat + DIMN, 2 * DIMN, CSZ);
    cvt_cat<<<cdiv(LSZ * 32, 256), 256, 0, stream>>>(c2lA, DIMN, l_cat + DIMN, 3 * DIMN, LSZ);

    // update linears -> next output slices (f32)
    gemm(c_cat, 2 * DIMN, 0, WtCU + (size_t)i * 2 * DIMN * DIMN, cu_b + i * DIMN,
         CSZ, 2 * DIMN, 0, c_next, DIMN, nullptr, 0);
    gemm(l_cat, 3 * DIMN, 0, WtLU + (size_t)i * 3 * DIMN * DIMN, lu_b + i * DIMN,
         LSZ, 3 * DIMN, 0, l_next, DIMN, nullptr, 0);
  }
}
